// RPN_1331439861972
// MI455X (gfx1250) — compile-verified, matrix operand order checked by reading, not run on hardware
//
#include <hip/hip_runtime.h>
#include <math.h>

typedef __attribute__((ext_vector_type(16))) __bf16 v16bf;
typedef __attribute__((ext_vector_type(8)))  __bf16 v8bf;
typedef __attribute__((ext_vector_type(8)))  float  v8f;

#define BATCH 32
#define CIN   512
#define COUT  512
#define FH    50
#define FW    37
#define HW    1850            // 50*37
#define KTOT  (CIN * 9)       // 4608 (ordered tap-major: k = tap*512 + cin)
#define NGT   20
#define NT128 ((HW + 127) / 128)   // 15 N-tiles for the big GEMM
#define NT64  ((HW + 63) / 64)     // 29 N-tiles for the head GEMM

// -------- CDNA5 async helpers (inline asm; ASYNCcnt-tracked LDS DMA) --------
__device__ __forceinline__ unsigned lds_off32(const void* p) {
    // generic pointers to LDS carry the in-allocation offset in the low 32 bits
    return (unsigned)(size_t)p;
}
__device__ __forceinline__ void async_load_b128(unsigned lds_addr, const void* gaddr) {
    asm volatile("global_load_async_to_lds_b128 %0, %1, off"
                 :: "v"(lds_addr), "v"(gaddr) : "memory");
}
__device__ __forceinline__ void wait_async0() {
    asm volatile("s_wait_asynccnt 0x0" ::: "memory");
}

// ---------------------------------------------------------------------------
// Kernel 0a: W3 [cout][cin][3][3] fp32 -> bf16 reordered to [cout][tap][cin].
// ---------------------------------------------------------------------------
__global__ void k_cvtW(const float* __restrict__ src, __bf16* __restrict__ dst)
{
    int idx = blockIdx.x * blockDim.x + threadIdx.x;
    if (idx >= COUT * KTOT) return;
    int k = idx % KTOT;
    int m = idx / KTOT;
    int ci = k / 9;
    int r  = k - ci * 9;               // tap = ky*3+kx
    dst[(size_t)m * KTOT + r * CIN + ci] = (__bf16)src[idx];
}

// ---------------------------------------------------------------------------
// Kernel 0b: feats NCHW fp32 -> NHWC bf16  (dst[b][p][c], coalesced writes).
// ---------------------------------------------------------------------------
__global__ void k_cvtF(const float* __restrict__ src, __bf16* __restrict__ dst)
{
    size_t idx = (size_t)blockIdx.x * blockDim.x + threadIdx.x;
    if (idx >= (size_t)BATCH * HW * CIN) return;
    int c = (int)(idx % CIN);
    size_t t = idx / CIN;
    int p = (int)(t % HW);
    int b = (int)(t / HW);
    dst[idx] = (__bf16)src[((size_t)b * CIN + c) * HW + p];
}

// ---------------------------------------------------------------------------
// Kernel 1: 3x3 conv + bias + ReLU as implicit GEMM via bf16 WMMA.
// Block: 256 threads (8 waves as 4x2). Tile: 128(M) x 128(N), K-step 32,
// double-buffered LDS; BOTH tiles staged with GLOBAL_LOAD_ASYNC_TO_LDS_B128
// (tap-major K => uniform tap per chunk; NHWC feats => contiguous B rows;
//  OOB taps redirected to a zeroed page). Wave: 2 A-frags x 4 B-frags = 8 WMMA.
// Output conv1 stored NHWC bf16 (vectorized 16B epilogue stores).
// ---------------------------------------------------------------------------
__global__ __launch_bounds__(256) void k_conv3x3(const __bf16* __restrict__ featsb,
                                                 const __bf16* __restrict__ W3r,
                                                 const float* __restrict__ b3,
                                                 const __bf16* __restrict__ zpage,
                                                 __bf16* __restrict__ conv1)
{
    __shared__ __bf16 As[2][128][32];   // [buf][m][k]   8KB each
    __shared__ __bf16 Bs[2][128][32];   // [buf][n][k]   8KB each (N-major)

    const int tid  = threadIdx.x;
    const int wave = tid >> 5;
    const int lane = tid & 31;
    const int wm   = wave >> 1;          // 0..3 : M sub-tile (32 rows)
    const int wn   = wave & 1;           // 0..1 : N sub-tile (64 cols)
    const int b    = blockIdx.z;
    const int m0   = blockIdx.y * 128;
    const int n0   = blockIdx.x * 128;
    const __bf16* fb = featsb + (size_t)b * HW * CIN;   // NHWC

    v8f acc[2][4] = {};

    const int sr = tid >> 1;             // 0..127 : staging row
    const int sk = (tid & 1) << 4;       // 0 or 16 : staging K-half
    const int p  = n0 + sr;              // spatial position for B tile
    const int py = p / FW;
    const int px = p - py * FW;
    const int ln = lane & 15;
    const int kb = (lane >> 4) << 3;     // 0 or 8 : per-lane K base (ISA layout)

    // ---- staging (all ASYNCcnt DMA) ----
    auto stageA = [&](int s, int buf) {
        const __bf16* g = W3r + (size_t)(m0 + sr) * KTOT + (s << 5) + sk;
        unsigned l = lds_off32(&As[buf][sr][sk]);
        async_load_b128(l,      (const void*)g);
        async_load_b128(l + 16, (const void*)(g + 8));
    };
    auto stageB = [&](int s, int buf) {
        const int r  = s >> 4;           // tap (uniform per chunk)
        const int c0 = (s & 15) << 5;    // cin base
        const int ky = r / 3;
        const int kx = r - ky * 3;
        const int yy = py + ky - 1;
        const int xx = px + kx - 1;
        const bool inb = (p < HW) & (yy >= 0) & (yy < FH) & (xx >= 0) & (xx < FW);
        const __bf16* g = inb ? (fb + (size_t)(yy * FW + xx) * CIN + c0 + sk) : zpage;
        unsigned l = lds_off32(&Bs[buf][sr][sk]);
        async_load_b128(l,      (const void*)g);
        async_load_b128(l + 16, (const void*)(g + 8));
    };
    auto load_frag = [&](const __bf16* row) {
        v16bf f;
        #pragma unroll
        for (int i = 0; i < 8; ++i) { f[i] = row[kb + i]; f[8 + i] = row[kb + 16 + i]; }
        return f;
    };

    // ---- prologue ----
    stageA(0, 0);
    stageB(0, 0);
    wait_async0();
    __syncthreads();

    const int NSTEPS = KTOT / 32;        // 144
    for (int s = 0; s < NSTEPS; ++s) {
        const int cur = s & 1;
        const int nxt = cur ^ 1;
        if (s + 1 < NSTEPS) {            // overlap next-tile DMA with compute
            stageA(s + 1, nxt);
            stageB(s + 1, nxt);
        }

        v16bf a0 = load_frag(&As[cur][wm * 32 + ln][0]);
        v16bf a1 = load_frag(&As[cur][wm * 32 + 16 + ln][0]);
        #pragma unroll
        for (int nt = 0; nt < 4; ++nt) {
            v16bf bf = load_frag(&Bs[cur][wn * 64 + nt * 16 + ln][0]);
            acc[0][nt] = __builtin_amdgcn_wmma_f32_16x16x32_bf16(
                false, a0, false, bf, (short)0, acc[0][nt], false, false);
            acc[1][nt] = __builtin_amdgcn_wmma_f32_16x16x32_bf16(
                false, a1, false, bf, (short)0, acc[1][nt], false, false);
        }

        wait_async0();                   // this wave's DMA into nxt has landed
        __syncthreads();                 // all waves done reading cur / filling nxt
    }

    // ---- epilogue: NHWC store, 8 consecutive M per lane -> one 16B store ----
    #pragma unroll
    for (int fi = 0; fi < 2; ++fi) {
        const int mbase = m0 + wm * 32 + fi * 16 + ((lane >> 4) << 3);
        #pragma unroll
        for (int nt = 0; nt < 4; ++nt) {
            const int n = n0 + wn * 64 + nt * 16 + ln;
            if (n < HW) {
                v8bf o;
                #pragma unroll
                for (int r = 0; r < 8; ++r) {
                    float v = acc[fi][nt][r] + b3[mbase + r];
                    v = v > 0.0f ? v : 0.0f;
                    o[r] = (__bf16)v;
                }
                *(v8bf*)(conv1 + ((size_t)b * HW + n) * COUT + mbase) = o;
            }
        }
    }
}

// ---------------------------------------------------------------------------
// Kernel 2: both 1x1 convs as one WMMA GEMM (conv1 is NHWC bf16).
// M=64 rows: 0..17 = Wcls, 18..53 = Wloc, 54..63 = zero pad. K=512, N=HW.
// B tile staged with async-to-LDS (contiguous NHWC rows).
// ---------------------------------------------------------------------------
__global__ __launch_bounds__(128) void k_heads(const __bf16* __restrict__ conv1,
                                               const float* __restrict__ Wcls,
                                               const float* __restrict__ bcls,
                                               const float* __restrict__ Wloc,
                                               const float* __restrict__ bloc,
                                               const __bf16* __restrict__ zpage,
                                               float* __restrict__ cls_ws,
                                               float* __restrict__ out_loc)
{
    __shared__ __bf16 As[64][32];
    __shared__ __bf16 Bs[64][32];

    const int tid  = threadIdx.x;
    const int wave = tid >> 5;
    const int lane = tid & 31;
    const int b    = blockIdx.z;
    const int n0   = blockIdx.x * 64;

    v8f acc[4] = {};

    const int ar = tid >> 1;
    const int ak = (tid & 1) << 4;
    const int p  = n0 + ar;
    const int fm = (wave << 4) + (lane & 15);
    const int kb = (lane >> 4) << 3;
    const __bf16* cb = conv1 + (size_t)b * HW * CIN;   // NHWC

    for (int k0 = 0; k0 < CIN; k0 += 32) {
        // B tile: contiguous 32 channels at position p -> async DMA
        {
            const __bf16* g = (p < HW) ? (cb + (size_t)p * CIN + k0 + ak) : zpage;
            unsigned l = lds_off32(&Bs[ar][ak]);
            async_load_b128(l,      (const void*)g);
            async_load_b128(l + 16, (const void*)(g + 8));
        }
        // A tile: small head weights, fp32 -> bf16 inline
        #pragma unroll
        for (int i = 0; i < 16; ++i) {
            int k = k0 + ak + i;
            float w = 0.0f;
            if (ar < 18)       w = Wcls[ar * CIN + k];
            else if (ar < 54)  w = Wloc[(ar - 18) * CIN + k];
            As[ar][ak + i] = (__bf16)w;
        }
        wait_async0();
        __syncthreads();

        v16bf afrag;
        #pragma unroll
        for (int i = 0; i < 8; ++i) {
            afrag[i]     = As[fm][kb + i];
            afrag[8 + i] = As[fm][kb + 16 + i];
        }
        #pragma unroll
        for (int nt = 0; nt < 4; ++nt) {
            const int fn = (nt << 4) + (lane & 15);
            v16bf bfrag;
            #pragma unroll
            for (int i = 0; i < 8; ++i) {
                bfrag[i]     = Bs[fn][kb + i];
                bfrag[8 + i] = Bs[fn][kb + 16 + i];
            }
            acc[nt] = __builtin_amdgcn_wmma_f32_16x16x32_bf16(
                false, afrag, false, bfrag, (short)0, acc[nt], false, false);
        }
        __syncthreads();
    }

    const int mbase = (wave << 4) + ((lane >> 4) << 3);
    #pragma unroll
    for (int nt = 0; nt < 4; ++nt) {
        const int n = n0 + (nt << 4) + (lane & 15);
        if (n < HW) {
            #pragma unroll
            for (int r = 0; r < 8; ++r) {
                const int m = mbase + r;
                float v = acc[nt][r];
                if (m < 18)
                    cls_ws[((size_t)b * 18 + m) * HW + n] = v + bcls[m];
                else if (m < 54)
                    out_loc[((size_t)b * 36 + (m - 18)) * HW + n] = v + bloc[m - 18];
            }
        }
    }
}

// ---------------------------------------------------------------------------
// Kernel 3: pairwise softmax over channels (c, c+9) -> prob_back in d_out.
// ---------------------------------------------------------------------------
__global__ void k_softmax(const float* __restrict__ cls_ws, float* __restrict__ prob)
{
    int idx = blockIdx.x * blockDim.x + threadIdx.x;
    if (idx >= BATCH * 9 * HW) return;
    int p = idx % HW;
    int t = idx / HW;
    int j = t % 9;
    int b = t / 9;
    float l0 = cls_ws[((size_t)b * 18 + j) * HW + p];
    float l1 = cls_ws[((size_t)b * 18 + 9 + j) * HW + p];
    float m  = fmaxf(l0, l1);
    float e0 = __expf(l0 - m), e1 = __expf(l1 - m);
    float s  = e0 + e1;
    prob[((size_t)b * 18 + j) * HW + p]     = e0 / s;
    prob[((size_t)b * 18 + 9 + j) * HW + p] = e1 / s;
}

// ---------------------------------------------------------------------------
// Kernel 4: zero per-call state: gt_best keys, loss scalars, async zero page.
// ---------------------------------------------------------------------------
__global__ void k_init(unsigned long long* __restrict__ gt_best,
                       float* __restrict__ loss, __bf16* __restrict__ zpage)
{
    int i = threadIdx.x;
    if (i < BATCH * NGT) gt_best[i] = 0ull;
    if (i < 128) zpage[i] = (__bf16)0.0f;       // 256B zero page
    if (i == 0) { loss[0] = 0.0f; loss[1] = 0.0f; }
}

// ---------------------------------------------------------------------------
// Kernel 5: per-anchor IoU vs 20 GT boxes: max/argmax over g, plus
// best-anchor-per-gt via packed u64 atomicMax (iou bits || ~a -> first max).
// ---------------------------------------------------------------------------
__global__ void k_iou(const float* __restrict__ anchors, const float* __restrict__ gt,
                      int A, float* __restrict__ max_iou, int* __restrict__ matched,
                      unsigned long long* __restrict__ gt_best)
{
    int a = blockIdx.x * blockDim.x + threadIdx.x;
    int b = blockIdx.y;
    if (a >= A) return;
    float ax1 = anchors[a * 4 + 0], ay1 = anchors[a * 4 + 1];
    float ax2 = anchors[a * 4 + 2], ay2 = anchors[a * 4 + 3];
    float area_a = (ax2 - ax1) * (ay2 - ay1);
    float best = -1.0f; int bg = 0;
    for (int g = 0; g < NGT; ++g) {
        const float* gb = gt + ((size_t)b * NGT + g) * 4;
        float ix1 = fmaxf(ax1, gb[0]);
        float iy1 = fmaxf(ay1, gb[1]);
        float ix2 = fminf(ax2, gb[2]);
        float iy2 = fminf(ay2, gb[3]);
        float inter = fmaxf(ix2 - ix1, 0.0f) * fmaxf(iy2 - iy1, 0.0f);
        float area_g = (gb[2] - gb[0]) * (gb[3] - gb[1]);
        float iou = inter / fmaxf(area_a + area_g - inter, 1e-8f);
        if (iou > best) { best = iou; bg = g; }
        unsigned long long key = ((unsigned long long)__float_as_uint(iou) << 32)
                               | (unsigned long long)(0xFFFFFFFFu - (unsigned)a);
        atomicMax(&gt_best[b * NGT + g], key);
    }
    max_iou[(size_t)b * A + a] = best;
    matched[(size_t)b * A + a] = bg;
}

// ---------------------------------------------------------------------------
// Kernel 6: labels + cls/loc losses (reference's asymmetric decompositions).
// ---------------------------------------------------------------------------
__global__ void k_loss(const float* __restrict__ cls_ws, const float* __restrict__ out_loc,
                       const float* __restrict__ gt, const int* __restrict__ idxs, int A,
                       const float* __restrict__ max_iou, const int* __restrict__ matched,
                       const unsigned long long* __restrict__ gt_best,
                       float* __restrict__ loss)
{
    int a = blockIdx.x * blockDim.x + threadIdx.x;
    int b = blockIdx.y;
    if (a >= A) return;

    bool best = false;
    for (int g = 0; g < NGT; ++g) {
        unsigned aw = 0xFFFFFFFFu - (unsigned)(gt_best[b * NGT + g] & 0xFFFFFFFFull);
        if ((int)aw == a) best = true;
    }
    int label = (best || max_iou[(size_t)b * A + a] > 0.7f) ? 1 : 0;
    int flat  = idxs[a];

    int j  = flat / HW;
    int pc = flat - j * HW;
    float l0 = cls_ws[((size_t)b * 18 + j) * HW + pc];
    float l1 = cls_ws[((size_t)b * 18 + 9 + j) * HW + pc];
    float m   = fmaxf(l0, l1);
    float lse = m + __logf(__expf(l0 - m) + __expf(l1 - m));
    float ll  = (label ? l1 : l0) - lse;
    atomicAdd(&loss[0], -ll / (float)(BATCH * A));

    if (label) {
        int q  = flat % 9;
        int pl = flat / 9;
        int mg = matched[(size_t)b * A + a];
        float s = 0.0f;
        #pragma unroll
        for (int c = 0; c < 4; ++c) {
            float lp = out_loc[((size_t)b * 36 + (4 * q + c)) * HW + pl];
            float tg = gt[((size_t)b * NGT + mg) * 4 + c];
            float d  = lp - tg;
            float ad = fabsf(d);
            s += (ad < 1.0f) ? 0.5f * d * d : ad - 0.5f;
        }
        atomicAdd(&loss[1], s / 256.0f);
    }
}

// ---------------------------------------------------------------------------
extern "C" void kernel_launch(void* const* d_in, const int* in_sizes, int n_in,
                              void* d_out, int out_size, void* d_ws, size_t ws_size,
                              hipStream_t stream)
{
    const float* feats   = (const float*)d_in[0];
    const float* gt      = (const float*)d_in[1];
    const float* anchors = (const float*)d_in[2];
    const float* W3      = (const float*)d_in[3];
    const float* b3      = (const float*)d_in[4];
    const float* Wcls    = (const float*)d_in[5];
    const float* bcls    = (const float*)d_in[6];
    const float* Wloc    = (const float*)d_in[7];
    const float* bloc    = (const float*)d_in[8];
    const int*   idxs    = (const int*)d_in[9];
    const int A = in_sizes[9];

    float* out      = (float*)d_out;
    float* out_prob = out;                                     // [B,18,HW]
    float* out_loc  = out + (size_t)BATCH * 18 * HW;           // [B,36,HW]
    float* out_loss = out + (size_t)BATCH * 18 * HW + (size_t)BATCH * 36 * HW; // [2]

    char* ws = (char*)d_ws;
    size_t off = 0;
    auto walloc = [&](size_t bytes) {
        char* q = ws + off;
        off += (bytes + 255) & ~(size_t)255;
        return q;
    };
    __bf16* featsb = (__bf16*)walloc((size_t)BATCH * HW * CIN * sizeof(__bf16)); // NHWC
    __bf16* W3r    = (__bf16*)walloc((size_t)COUT * KTOT * sizeof(__bf16));      // tap-major
    __bf16* conv1  = (__bf16*)walloc((size_t)BATCH * HW * COUT * sizeof(__bf16));// NHWC
    float*  cls_ws = (float*)walloc((size_t)BATCH * 18 * HW * sizeof(float));
    float*  max_iou = (float*)walloc((size_t)BATCH * A * sizeof(float));
    int*    matched = (int*)walloc((size_t)BATCH * A * sizeof(int));
    unsigned long long* gt_best =
        (unsigned long long*)walloc((size_t)BATCH * NGT * sizeof(unsigned long long));
    __bf16* zpage = (__bf16*)walloc(256);                      // async OOB target

    // init first: zero page must be ready before the GEMM's async loads
    k_init<<<1, 1024, 0, stream>>>(gt_best, out_loss, zpage);

    // prep: layout-transforming fp32 -> bf16 conversions
    {
        int nw = COUT * KTOT;                    // 2,359,296
        k_cvtW<<<(nw + 255) / 256, 256, 0, stream>>>(W3, W3r);
        size_t nf = (size_t)BATCH * HW * CIN;    // 30,310,400
        k_cvtF<<<(unsigned)((nf + 255) / 256), 256, 0, stream>>>(feats, featsb);
    }

    dim3 g1(NT128, COUT / 128, BATCH);
    k_conv3x3<<<g1, 256, 0, stream>>>(featsb, W3r, b3, zpage, conv1);

    dim3 g2(NT64, 1, BATCH);
    k_heads<<<g2, 128, 0, stream>>>(conv1, Wcls, bcls, Wloc, bloc, zpage, cls_ws, out_loc);

    int nsm = BATCH * 9 * HW;
    k_softmax<<<(nsm + 255) / 256, 256, 0, stream>>>(cls_ws, out_prob);

    dim3 g4((A + 255) / 256, BATCH, 1);
    k_iou<<<g4, 256, 0, stream>>>(anchors, gt, A, max_iou, matched, gt_best);
    k_loss<<<g4, 256, 0, stream>>>(cls_ws, out_loc, gt, idxs, A, max_iou, matched,
                                   gt_best, out_loss);
}
